// LAEF_43611097923793
// MI455X (gfx1250) — matrix operand order repaired and checked by reading, not compiled, not run on hardware
//
#include <hip/hip_runtime.h>
#include <cstdint>
#include <cstddef>

// ---------------------------------------------------------------------------
// LAEF fusion module for MI455X (gfx1250, wave32, WMMA).
// Heavy convs (w_off1, DCN einsum, w_f) run as im2col GEMMs on
// v_wmma_f32_16x16x32_f16; activations staged as padded f16 NHWC.
// ---------------------------------------------------------------------------

typedef _Float16 f16;
typedef __attribute__((ext_vector_type(8)))  _Float16 v8h;
typedef __attribute__((ext_vector_type(16))) _Float16 v16h;
typedef __attribute__((ext_vector_type(8)))  float    v8f;
typedef __attribute__((ext_vector_type(4)))  float    v4f;

#define NB   8      // batch
#define CCH  128    // channels per stream
#define HH   80
#define WW   80
#define HP   82     // padded
#define WP   82
#define MIDC 16
#define K1TOT 2304  // 256*9
#define KDTOT 1152  // 128*9

__device__ __forceinline__ float siluf(float v) { return v / (1.0f + __expf(-v)); }
__device__ __forceinline__ float sigmf(float v) { return 1.0f / (1.0f + __expf(-v)); }

// One 16x16x32 WMMA step. aPtr: LDS row base for this lane (half offset already
// applied); bPtr: packed-weight fragment base (16 contiguous f16 K for lane's N).
__device__ __forceinline__ v8f wmma_step(const f16* aPtr, const f16* bPtr, v8f acc) {
    v8h a_lo = *(const v8h*)aPtr;          // K 0..7  (or 16..23 for upper half)
    v8h a_hi = *(const v8h*)(aPtr + 8);    // K 8..15 (or 24..31)
    v16h aF = __builtin_shufflevector(a_lo, a_hi, 0,1,2,3,4,5,6,7,8,9,10,11,12,13,14,15);
    v16h bF = *(const v16h*)bPtr;
#if defined(__gfx1250__)
    acc = __builtin_amdgcn_wmma_f32_16x16x32_f16(false, aF, false, bF, (short)0, acc,
                                                 false, false);
#else
    acc[0] += (float)aF[0] * (float)bF[0]; // host-pass placeholder
#endif
    return acc;
}

// ---------------------------------------------------------------------------
// Utility / packing kernels
// ---------------------------------------------------------------------------
__global__ void zero_u32_kernel(uint32_t* p, size_t n32) {
    size_t i = (size_t)blockIdx.x * blockDim.x + threadIdx.x;
    size_t stride = (size_t)gridDim.x * blockDim.x;
    for (; i < n32; i += stride) p[i] = 0u;
}

// fp32 NCHW rgb|ir -> padded f16 NHWC cat_pad [B][82][82][256] (border pre-zeroed)
__global__ void pack_cat_kernel(const float* __restrict__ rgb,
                                const float* __restrict__ ir,
                                f16* __restrict__ catPad) {
    size_t t = (size_t)blockIdx.x * 256 + threadIdx.x;
    size_t total = (size_t)NB * 256 * HH * WW;
    if (t >= total) return;
    int x = (int)(t % WW);
    int y = (int)((t / WW) % HH);
    int c = (int)((t / (WW * HH)) % 256);
    int b = (int)(t / ((size_t)WW * HH * 256));
    float v = (c < CCH)
        ? rgb[(((size_t)b * CCH + c) * HH + y) * WW + x]
        : ir [(((size_t)b * CCH + (c - CCH)) * HH + y) * WW + x];
    catPad[((size_t)((b * HP + (y + 1)) * WP + (x + 1))) * 256 + c] = (f16)v;
}

// OIHW fp32 -> [N][K] f16, K = (ky*3+kx)*Cin + c. Optional BN-gamma/var scale fold.
__global__ void pack_w_kernel(const float* __restrict__ w, f16* __restrict__ wp,
                              int N, int Cin, const float* __restrict__ bn) {
    int Kt = Cin * 9;
    long t = (long)blockIdx.x * 256 + threadIdx.x;
    if (t >= (long)N * Kt) return;
    int n = (int)(t / Kt);
    int k = (int)(t % Kt);
    int kpos = k / Cin;
    int c    = k % Cin;
    float v = w[((size_t)n * Cin + c) * 9 + kpos];
    if (bn) v *= bn[n] * rsqrtf(bn[3 * N + n] + 1e-5f);
    wp[(size_t)n * Kt + k] = (f16)v;
}

// ---------------------------------------------------------------------------
// Big 3x3 conv as im2col GEMM with WMMA. Cin=256, Ktot=2304, Cout=128.
// mode 0: +bias, SiLU, store padded f16 (h buffer)
// mode 1: +folded-BN shift, SiLU, +res_scale*ir_aligned, store fp32 NCHW d_out
// ---------------------------------------------------------------------------
__global__ __launch_bounds__(256)
void gemm_conv3x3_kernel(const f16* __restrict__ srcPad,     // [B][82][82][256]
                         const f16* __restrict__ wPack,      // [128][2304]
                         const float* __restrict__ bias,     // mode0
                         const float* __restrict__ bnf,      // mode1: [4][128]
                         const float* __restrict__ irAligned,// mode1: NHWC fp32
                         const float* __restrict__ resScale, // mode1: scalar
                         f16* __restrict__ dstPad,           // mode0: [B][82][82][128]
                         float* __restrict__ outNCHW,        // mode1
                         int mode) {
    __shared__ f16 lds_a[16 * 136];   // 16 rows x 128 cols, +8 pad vs banks

    int mt = blockIdx.x;              // 3200 = B*H*(W/16)
    int x0 = (mt % 5) * 16;
    int t2 = mt / 5;
    int y  = t2 % HH;
    int b  = t2 / HH;

    int tid  = threadIdx.x;
    int lane = tid & 31;
    int wv   = tid >> 5;              // 8 waves -> 8 N-tiles of 16
    int half = lane >> 4;             // 0: lanes 0-15, 1: lanes 16-31
    int nn   = wv * 16 + (lane & 15); // output channel for B/C fragments
    int lm   = tid >> 4;              // A-tile load row
    int lc   = (tid & 15) * 8;        // A-tile load col (8 f16 per thread)

    v8f acc = {};
    for (int kc = 0; kc < 18; ++kc) { // 18 * 128 = 2304
        int base_k = kc * 128;
        int kpos   = kc >> 1;
        int coff   = (kc & 1) * 128;
        int ky = kpos / 3, kx = kpos % 3;

        const f16* srow = srcPad +
            ((size_t)((b * HP + (y + ky)) * WP + (x0 + lm + kx))) * 256 + coff + lc;
        v8h av = *(const v8h*)srow;

        __builtin_prefetch(wPack + (size_t)nn * K1TOT + base_k + 128, 0, 1);

        __syncthreads();                       // previous iter's A reads done
        *(v8h*)&lds_a[lm * 136 + lc] = av;
        __syncthreads();

#pragma unroll
        for (int q = 0; q < 4; ++q) {
            const f16* aPtr = &lds_a[(lane & 15) * 136 + q * 32 + half * 16];
            const f16* bPtr = wPack + (size_t)nn * K1TOT + base_k + q * 32 + half * 16;
            acc = wmma_step(aPtr, bPtr, acc);
        }
    }

    int mbase = half * 8;
    if (mode == 0) {
        float bv = bias[nn];
#pragma unroll
        for (int i = 0; i < 8; ++i) {
            float v = acc[i] + bv;
            int xo = x0 + mbase + i;
            dstPad[((size_t)((b * HP + (y + 1)) * WP + (xo + 1))) * CCH + nn] =
                (f16)siluf(v);
        }
    } else {
        float gamma = bnf[nn], beta = bnf[CCH + nn];
        float mean  = bnf[2 * CCH + nn], var = bnf[3 * CCH + nn];
        float scl   = gamma * rsqrtf(var + 1e-5f);   // already folded into weights
        float shift = beta - mean * scl;
        float rs    = resScale[0];
#pragma unroll
        for (int i = 0; i < 8; ++i) {
            int xo = x0 + mbase + i;
            float v   = acc[i] + shift;
            float ira = irAligned[((size_t)((b * HH + y) * WW + xo)) * CCH + nn];
            outNCHW[(((size_t)b * CCH + nn) * HH + y) * WW + xo] = siluf(v) + rs * ira;
        }
    }
}

// ---------------------------------------------------------------------------
// Offset/mask conv: 128->27, 3x3, direct VALU (3.2 GFLOP).
// offs layout per pixel: [dy(9) | dx(9) | sigmoid(mask)(9)]
// ---------------------------------------------------------------------------
__global__ __launch_bounds__(256)
void offset_conv_kernel(const f16* __restrict__ hPad,   // [B][82][82][128]
                        const f16* __restrict__ wp2,    // [27][1152]
                        const float* __restrict__ b_off2,
                        float* __restrict__ offs) {
    int pix = blockIdx.x * 256 + threadIdx.x;
    if (pix >= NB * HH * WW) return;
    int x = pix % WW;
    int y = (pix / WW) % HH;
    int b = pix / (WW * HH);

    float acc[27];
#pragma unroll
    for (int co = 0; co < 27; ++co) acc[co] = b_off2[co];

    for (int kpos = 0; kpos < 9; ++kpos) {
        int ky = kpos / 3, kx = kpos % 3;
        const f16* src = hPad + ((size_t)((b * HP + (y + ky)) * WP + (x + kx))) * CCH;
        for (int c = 0; c < CCH; c += 8) {
            v8h hv = *(const v8h*)(src + c);
#pragma unroll
            for (int j = 0; j < 8; ++j) {
                float h = (float)hv[j];
                const f16* wcol = wp2 + (size_t)kpos * CCH + c + j;
#pragma unroll
                for (int co = 0; co < 27; ++co)
                    acc[co] += h * (float)wcol[(size_t)co * KDTOT];
            }
        }
    }
    size_t ob = (size_t)pix * 27;
#pragma unroll
    for (int k = 0; k < 9; ++k) {
        offs[ob + k]      = acc[2 * k];           // dy
        offs[ob + 9 + k]  = acc[2 * k + 1];       // dx
        offs[ob + 18 + k] = sigmf(acc[18 + k]);   // mask
    }
}

// ---------------------------------------------------------------------------
// Deformable sampling fused with DCN GEMM (K = 9*128) on WMMA.
// ---------------------------------------------------------------------------
__global__ __launch_bounds__(256)
void dcn_kernel(const f16* __restrict__ catPad,      // ir = channels 128..255
                const f16* __restrict__ wpd,         // [128][1152]
                const float* __restrict__ offs,      // [pix][27]
                const float* __restrict__ b_dcn,
                float* __restrict__ irAligned) {     // NHWC fp32
    __shared__ f16   lds_a[16 * 136];
    __shared__ float s_w[16][4];
    __shared__ int   s_ad[16][4];

    int mt = blockIdx.x;
    int x0 = (mt % 5) * 16;
    int t2 = mt / 5;
    int y  = t2 % HH;
    int b  = t2 / HH;

    int tid  = threadIdx.x;
    int lane = tid & 31;
    int wv   = tid >> 5;
    int half = lane >> 4;
    int nn   = wv * 16 + (lane & 15);

    v8f acc = {};
    for (int kpos = 0; kpos < 9; ++kpos) {
        __syncthreads();                       // previous tile fully consumed
        if (tid < 16) {
            int m = tid, x = x0 + m;
            size_t pb = ((size_t)(b * HH + y) * WW + x) * 27;
            float dy  = offs[pb + kpos];
            float dx  = offs[pb + 9 + kpos];
            float msk = offs[pb + 18 + kpos];
            float fy = (float)y + (float)(kpos / 3 - 1) + dy;
            float fx = (float)x + (float)(kpos % 3 - 1) + dx;
            float y0f = floorf(fy), x0f = floorf(fx);
            float wy = fy - y0f, wx = fx - x0f;
            int yi0 = (int)y0f, xi0 = (int)x0f;
            int yi1 = yi0 + 1,  xi1 = xi0 + 1;
            float vy0 = (yi0 >= 0 && yi0 < HH) ? 1.f : 0.f;
            float vy1 = (yi1 >= 0 && yi1 < HH) ? 1.f : 0.f;
            float vx0 = (xi0 >= 0 && xi0 < WW) ? 1.f : 0.f;
            float vx1 = (xi1 >= 0 && xi1 < WW) ? 1.f : 0.f;
            s_w[m][0] = (1 - wy) * (1 - wx) * msk * vy0 * vx0;
            s_w[m][1] = (1 - wy) * wx       * msk * vy0 * vx1;
            s_w[m][2] = wy       * (1 - wx) * msk * vy1 * vx0;
            s_w[m][3] = wy       * wx       * msk * vy1 * vx1;
            int yc0 = min(max(yi0, 0), HH - 1), yc1 = min(max(yi1, 0), HH - 1);
            int xc0 = min(max(xi0, 0), WW - 1), xc1 = min(max(xi1, 0), WW - 1);
            s_ad[m][0] = ((b * HP + yc0 + 1) * WP + xc0 + 1) * 256 + CCH;
            s_ad[m][1] = ((b * HP + yc0 + 1) * WP + xc1 + 1) * 256 + CCH;
            s_ad[m][2] = ((b * HP + yc1 + 1) * WP + xc0 + 1) * 256 + CCH;
            s_ad[m][3] = ((b * HP + yc1 + 1) * WP + xc1 + 1) * 256 + CCH;
        }
        __syncthreads();

        { // gather + blend 8 channels per thread into A tile
            int m = tid >> 4;
            int c = (tid & 15) * 8;
            float av[8] = {0, 0, 0, 0, 0, 0, 0, 0};
#pragma unroll
            for (int p = 0; p < 4; ++p) {
                float wp = s_w[m][p];
                v8h v = *(const v8h*)(catPad + s_ad[m][p] + c);
#pragma unroll
                for (int j = 0; j < 8; ++j) av[j] += wp * (float)v[j];
            }
            v8h st;
#pragma unroll
            for (int j = 0; j < 8; ++j) st[j] = (f16)av[j];
            *(v8h*)&lds_a[m * 136 + c] = st;
        }
        __syncthreads();

#pragma unroll
        for (int q = 0; q < 4; ++q) {
            const f16* aPtr = &lds_a[(lane & 15) * 136 + q * 32 + half * 16];
            const f16* bPtr = wpd + (size_t)nn * KDTOT + kpos * 128 + q * 32 + half * 16;
            acc = wmma_step(aPtr, bPtr, acc);
        }
    }

    float bv = b_dcn[nn];
#pragma unroll
    for (int i = 0; i < 8; ++i) {
        int xo = x0 + half * 8 + i;
        irAligned[((size_t)((b * HH + y) * WW + xo)) * CCH + nn] = acc[i] + bv;
    }
}

// ---------------------------------------------------------------------------
// Gating path: 1x1 (256->16) +BN+SiLU, depthwise 3x3 +BN+SiLU, 1x1 (16->1)+sigmoid
// One block per 16x16 spatial tile (18x18 halo of the 16-ch map in LDS).
// ---------------------------------------------------------------------------
__global__ __launch_bounds__(256)
void gate_kernel(const f16* __restrict__ catPad,      // rgb = channels 0..127
                 const float* __restrict__ irAligned, // NHWC fp32
                 const float* __restrict__ w_g1, const float* __restrict__ bn_g1,
                 const float* __restrict__ w_g2, const float* __restrict__ bn_g2,
                 const float* __restrict__ w_g3, const float* __restrict__ b_g3,
                 float* __restrict__ gOut) {
    __shared__ float s_w1[MIDC * 256];
    __shared__ float s1[18 * 18 * MIDC];
    __shared__ float s_sc1[MIDC], s_sh1[MIDC], s_sc2[MIDC], s_sh2[MIDC], s_w3[MIDC];

    int blk = blockIdx.x;             // 200 = b*25 + ty*5 + tx
    int tx = blk % 5, t2 = blk / 5;
    int ty = t2 % 5, b = t2 / 5;
    int X0 = tx * 16, Y0 = ty * 16;
    int tid = threadIdx.x;

    for (int i = tid; i < MIDC * 256; i += 256) s_w1[i] = w_g1[i];
    if (tid < MIDC) {
        float g = bn_g1[tid], be = bn_g1[MIDC + tid];
        float mn = bn_g1[2 * MIDC + tid], vr = bn_g1[3 * MIDC + tid];
        float sc = g * rsqrtf(vr + 1e-5f);
        s_sc1[tid] = sc; s_sh1[tid] = be - mn * sc;
        g = bn_g2[tid]; be = bn_g2[MIDC + tid];
        mn = bn_g2[2 * MIDC + tid]; vr = bn_g2[3 * MIDC + tid];
        sc = g * rsqrtf(vr + 1e-5f);
        s_sc2[tid] = sc; s_sh2[tid] = be - mn * sc;
        s_w3[tid] = w_g3[tid];
    }
    __syncthreads();

    for (int p = tid; p < 18 * 18; p += 256) {
        int hy = p / 18, hx = p % 18;
        int yy = Y0 + hy - 1, xx = X0 + hx - 1;
        float a[MIDC];
#pragma unroll
        for (int ch = 0; ch < MIDC; ++ch) a[ch] = 0.f;
        if (yy >= 0 && yy < HH && xx >= 0 && xx < WW) {
            const f16* rgbp = catPad + ((size_t)((b * HP + yy + 1) * WP + (xx + 1))) * 256;
            const float* irp = irAligned + ((size_t)(b * HH + yy) * WW + xx) * CCH;
            for (int c = 0; c < CCH; c += 8) {
                v8h r = *(const v8h*)(rgbp + c);
#pragma unroll
                for (int j = 0; j < 8; ++j) {
                    float iv = (float)r[j];
#pragma unroll
                    for (int ch = 0; ch < MIDC; ++ch)
                        a[ch] += iv * s_w1[ch * 256 + c + j];
                }
            }
            for (int c = 0; c < CCH; c += 4) {
                v4f ir4 = *(const v4f*)(irp + c);
#pragma unroll
                for (int j = 0; j < 4; ++j) {
                    float iv = ir4[j];
#pragma unroll
                    for (int ch = 0; ch < MIDC; ++ch)
                        a[ch] += iv * s_w1[ch * 256 + CCH + c + j];
                }
            }
#pragma unroll
            for (int ch = 0; ch < MIDC; ++ch) {
                float v = a[ch] * s_sc1[ch] + s_sh1[ch];
                s1[p * MIDC + ch] = siluf(v);
            }
        } else {
#pragma unroll
            for (int ch = 0; ch < MIDC; ++ch) s1[p * MIDC + ch] = 0.f;
        }
    }
    __syncthreads();

    int py = tid >> 4, px = tid & 15;
    float gacc = b_g3[0];
#pragma unroll
    for (int ch = 0; ch < MIDC; ++ch) {
        float d = 0.f;
#pragma unroll
        for (int dy = 0; dy < 3; ++dy)
#pragma unroll
            for (int dx = 0; dx < 3; ++dx)
                d += s1[((py + dy) * 18 + (px + dx)) * MIDC + ch] *
                     w_g2[ch * 9 + dy * 3 + dx];
        float v = d * s_sc2[ch] + s_sh2[ch];
        gacc += siluf(v) * s_w3[ch];
    }
    gOut[(size_t)(b * HH + Y0 + py) * WW + X0 + px] = sigmf(gacc);
}

// ---------------------------------------------------------------------------
// fused_in = concat(g*ir_aligned, (1-g)*rgb) -> padded f16 NHWC [B][82][82][256]
// ---------------------------------------------------------------------------
__global__ __launch_bounds__(256)
void fuse_build_kernel(const f16* __restrict__ catPad,
                       const float* __restrict__ irAligned,
                       const float* __restrict__ gOut,
                       f16* __restrict__ fusedPad) {
    long t = (long)blockIdx.x * 256 + threadIdx.x;
    long total = (long)NB * HH * WW * 32;   // 8 channels per thread
    if (t >= total) return;
    int cg  = (int)(t & 31);
    int pix = (int)(t >> 5);
    int x = pix % WW;
    int y = (pix / WW) % HH;
    int b = pix / (WW * HH);
    int c0 = cg * 8;
    float gv = gOut[pix];
    size_t padPix = (size_t)((b * HP + (y + 1)) * WP + (x + 1));
    v8h out;
    if (c0 < CCH) {
        const float* irp = irAligned + (size_t)pix * CCH + c0;
#pragma unroll
        for (int j = 0; j < 8; ++j) out[j] = (f16)(gv * irp[j]);
    } else {
        v8h r = *(const v8h*)(catPad + padPix * 256 + (c0 - CCH));
        float og = 1.f - gv;
#pragma unroll
        for (int j = 0; j < 8; ++j) out[j] = (f16)(og * (float)r[j]);
    }
    *(v8h*)(fusedPad + padPix * 256 + c0) = out;
}

// ---------------------------------------------------------------------------
// Host-side launch
// ---------------------------------------------------------------------------
extern "C" void kernel_launch(void* const* d_in, const int* in_sizes, int n_in,
                              void* d_out, int out_size, void* d_ws, size_t ws_size,
                              hipStream_t stream) {
    const float* rgb      = (const float*)d_in[0];
    const float* ir       = (const float*)d_in[1];
    const float* w_off1   = (const float*)d_in[2];
    const float* b_off1   = (const float*)d_in[3];
    const float* w_off2   = (const float*)d_in[4];
    const float* b_off2   = (const float*)d_in[5];
    const float* w_dcn    = (const float*)d_in[6];
    const float* b_dcn    = (const float*)d_in[7];
    const float* w_g1     = (const float*)d_in[8];
    const float* bn_g1    = (const float*)d_in[9];
    const float* w_g2     = (const float*)d_in[10];
    const float* bn_g2    = (const float*)d_in[11];
    const float* w_g3     = (const float*)d_in[12];
    const float* b_g3     = (const float*)d_in[13];
    const float* w_f      = (const float*)d_in[14];
    const float* bn_f     = (const float*)d_in[15];
    const float* resScale = (const float*)d_in[16];

    // Workspace carve (256B aligned). Zero-region buffers first & contiguous.
    uint8_t* ws = (uint8_t*)d_ws;
    size_t off = 0;
    auto carve = [&](size_t bytes) {
        size_t o = off;
        off = (off + bytes + 255) & ~(size_t)255;
        return o;
    };
    const size_t catPadB   = (size_t)NB * HP * WP * 256 * sizeof(f16);
    const size_t hPadB     = (size_t)NB * HP * WP * CCH * sizeof(f16);
    const size_t fusedPadB = catPadB;
    size_t o_cat   = carve(catPadB);
    size_t o_h     = carve(hPadB);
    size_t o_fused = carve(fusedPadB);
    size_t zeroEnd = off;                         // zero the padded f16 buffers
    size_t o_offs  = carve((size_t)NB * HH * WW * 27 * sizeof(float));
    size_t o_iral  = carve((size_t)NB * HH * WW * CCH * sizeof(float));
    size_t o_g     = carve((size_t)NB * HH * WW * sizeof(float));
    size_t o_wp1   = carve((size_t)CCH * K1TOT * sizeof(f16));
    size_t o_wp2   = carve((size_t)27 * KDTOT * sizeof(f16));
    size_t o_wpd   = carve((size_t)CCH * KDTOT * sizeof(f16));
    size_t o_wpf   = carve((size_t)CCH * K1TOT * sizeof(f16));
    (void)ws_size; (void)n_in; (void)in_sizes; (void)out_size;

    f16*   catPad   = (f16*)(ws + o_cat);
    f16*   hPad     = (f16*)(ws + o_h);
    f16*   fusedPad = (f16*)(ws + o_fused);
    float* offs     = (float*)(ws + o_offs);
    float* irAl     = (float*)(ws + o_iral);
    float* gBuf     = (float*)(ws + o_g);
    f16*   wp1      = (f16*)(ws + o_wp1);
    f16*   wp2      = (f16*)(ws + o_wp2);
    f16*   wpd      = (f16*)(ws + o_wpd);
    f16*   wpf      = (f16*)(ws + o_wpf);
    float* out      = (float*)d_out;

    // 0) zero padded staging buffers (borders must read as 0)
    size_t n32 = zeroEnd / 4;
    zero_u32_kernel<<<4096, 256, 0, stream>>>((uint32_t*)ws, n32);

    // 1) pack activations & weights
    {
        long total = (long)NB * 256 * HH * WW;
        pack_cat_kernel<<<(unsigned)((total + 255) / 256), 256, 0, stream>>>(rgb, ir, catPad);
    }
    pack_w_kernel<<<(128 * K1TOT + 255) / 256, 256, 0, stream>>>(w_off1, wp1, 128, 256, nullptr);
    pack_w_kernel<<<(27 * KDTOT + 255) / 256, 256, 0, stream>>>(w_off2, wp2, 27, 128, nullptr);
    pack_w_kernel<<<(128 * KDTOT + 255) / 256, 256, 0, stream>>>(w_dcn, wpd, 128, 128, nullptr);
    pack_w_kernel<<<(128 * K1TOT + 255) / 256, 256, 0, stream>>>(w_f, wpf, 128, 256, bn_f);

    const unsigned MTILES = NB * HH * (WW / 16);  // 3200

    // 2) h = silu(conv3x3(cat, w_off1) + b_off1)   [WMMA GEMM]
    gemm_conv3x3_kernel<<<MTILES, 256, 0, stream>>>(
        catPad, wp1, b_off1, nullptr, nullptr, nullptr, hPad, nullptr, 0);

    // 3) offsets + sigmoid(mask)
    offset_conv_kernel<<<(NB * HH * WW + 255) / 256, 256, 0, stream>>>(
        hPad, wp2, b_off2, offs);

    // 4) deformable sample + DCN GEMM -> ir_aligned  [WMMA]
    dcn_kernel<<<MTILES, 256, 0, stream>>>(catPad, wpd, offs, b_dcn, irAl);

    // 5) gating map g
    gate_kernel<<<NB * 25, 256, 0, stream>>>(catPad, irAl, w_g1, bn_g1,
                                             w_g2, bn_g2, w_g3, b_g3, gBuf);

    // 6) fused_in (padded f16 NHWC)
    {
        long total = (long)NB * HH * WW * 32;
        fuse_build_kernel<<<(unsigned)((total + 255) / 256), 256, 0, stream>>>(
            catPad, irAl, gBuf, fusedPad);
    }

    // 7) out = silu(bn(conv3x3(fused_in, w_f))) + res_scale*ir_aligned  [WMMA GEMM]
    gemm_conv3x3_kernel<<<MTILES, 256, 0, stream>>>(
        fusedPad, wpf, nullptr, bn_f, irAl, resScale, nullptr, out, 1);
}
